// TransformerBlock_68410239090890
// MI455X (gfx1250) — matrix operand order repaired and checked by reading, not compiled, not run on hardware
//
#include <hip/hip_runtime.h>
#include <hip/hip_bf16.h>

// ---------------------------------------------------------------------------
// Complex differential attention block for MI455X (gfx1250, wave32, WMMA).
// GEMMs on v_wmma_f32_16x16x32_f16; attention q-tiles staged in LDS via the
// CDNA5 async global->LDS path (ASYNCcnt); softmax in LDS (ds ops).
// ---------------------------------------------------------------------------

constexpr int kB = 2, kH = 8, kS = 1024, kD = 384;
constexpr int kM  = kB * kH * kS;   // 16384 rows
constexpr int kD2 = 2 * kD;         // 768
constexpr float kLambdaInit = 0.35550907f;          // 0.8 - 0.6*exp(-0.3)
constexpr float kScale      = 0.05103103630f;       // D^-0.5

typedef __attribute__((ext_vector_type(16))) _Float16 v16h;
typedef __attribute__((ext_vector_type(8)))  _Float16 v8h;
typedef __attribute__((ext_vector_type(8)))  float    v8f;

static __device__ __forceinline__ v8f wmma16(const v16h a, const v16h b, v8f c) {
  // D = A(16x32 f16) * B(32x16 f16) + C(16x16 f32)
  return __builtin_amdgcn_wmma_f32_16x16x32_f16(false, a, false, b, (short)0, c,
                                                false, false);
}

static __device__ __forceinline__ v16h neg16(v16h a) {
  v16h r;
#pragma unroll
  for (int j = 0; j < 16; ++j) r[j] = -a[j];
  return r;
}

// A-matrix fragment (16x32, MxK) from row-major f32 [*, ld]:
// lane m = lane&15, h = lane>>4; elems 0..7 -> K = k0+8h.. ; 8..15 -> K = k0+16+8h..
static __device__ __forceinline__ v16h a_frag_f32(const float* base, int ld,
                                                  int row, int k0, int h) {
  const float* p = base + (size_t)row * ld + k0 + 8 * h;
  float4 f0 = *(const float4*)(p);
  float4 f1 = *(const float4*)(p + 4);
  float4 f2 = *(const float4*)(p + 16);
  float4 f3 = *(const float4*)(p + 20);
  v16h a;
  a[0]=(_Float16)f0.x; a[1]=(_Float16)f0.y; a[2]=(_Float16)f0.z; a[3]=(_Float16)f0.w;
  a[4]=(_Float16)f1.x; a[5]=(_Float16)f1.y; a[6]=(_Float16)f1.z; a[7]=(_Float16)f1.w;
  a[8]=(_Float16)f2.x; a[9]=(_Float16)f2.y; a[10]=(_Float16)f2.z; a[11]=(_Float16)f2.w;
  a[12]=(_Float16)f3.x; a[13]=(_Float16)f3.y; a[14]=(_Float16)f3.z; a[15]=(_Float16)f3.w;
  return a;
}

static __device__ __forceinline__ v16h a_frag_f16(const _Float16* base, int ld,
                                                  int row, int k0, int h) {
  const _Float16* p = base + (size_t)row * ld + k0 + 8 * h;
  v8h lo = *(const v8h*)p;
  v8h hi = *(const v8h*)(p + 16);
  v16h a;
#pragma unroll
  for (int j = 0; j < 8; ++j) { a[j] = lo[j]; a[8 + j] = hi[j]; }
  return a;
}

// B-matrix fragment (32x16, KxN) where storage is [N, K] row-major f16:
// lane n = lane&15; elems 0..7 -> K = k0+8h.. ; 8..15 -> K = k0+16+8h..
static __device__ __forceinline__ v16h b_frag_f16(const _Float16* base, int ld,
                                                  int col, int k0, int h) {
  const _Float16* p = base + (size_t)col * ld + k0 + 8 * h;
  v8h lo = *(const v8h*)p;
  v8h hi = *(const v8h*)(p + 16);
  v16h b;
#pragma unroll
  for (int j = 0; j < 8; ++j) { b[j] = lo[j]; b[8 + j] = hi[j]; }
  return b;
}

// CDNA5 async global->LDS copy of one 16-byte chunk (tracked on ASYNCcnt).
static __device__ __forceinline__ void async_copy_b128(unsigned lds_addr,
                                                       const void* gaddr) {
  asm volatile("global_load_async_to_lds_b128 %0, %1, off"
               :
               : "v"(lds_addr), "v"((unsigned long long)(size_t)gaddr)
               : "memory");
}

static __device__ __forceinline__ void wait_asynccnt0() {
  asm volatile("s_wait_asynccnt 0" ::: "memory");
}

// ---------------------------------------------------------------------------
// Generic complex GEMM:  O = (Ar + i Ai) @ (Wr + i Wi)^T + (br + i bi) [+ PE]
// A: f32 [M, K]  W: f16 [N, K]  grid = (M/16, N/64), block = 128 (4 waves).
// mode 0: f32 out [M,N];  mode 1: f16 out [M,N] (+PE);  mode 2: f16 out [bh,N,S]
// ---------------------------------------------------------------------------
__global__ void cgemm_kernel(const float* __restrict__ Ar,
                             const float* __restrict__ Ai,
                             const _Float16* __restrict__ Wr,
                             const _Float16* __restrict__ Wi,
                             const float* __restrict__ br,
                             const float* __restrict__ bi,
                             int N, int K,
                             const float* __restrict__ per,
                             const float* __restrict__ pei,
                             int mode,
                             float* __restrict__ Orf, float* __restrict__ Oif,
                             _Float16* __restrict__ Orh,
                             _Float16* __restrict__ Oih) {
  const int wave = threadIdx.x >> 5;
  const int lane = threadIdx.x & 31;
  const int h    = lane >> 4;
  const int l15  = lane & 15;
  const int m0   = blockIdx.x * 16;
  const int ncol = blockIdx.y * 64 + wave * 16 + l15;
  const int mrow = m0 + l15;

  v8f accr = {};
  v8f acci = {};
  for (int k0 = 0; k0 < K; k0 += 32) {
    if (k0 + 32 < K) {   // prefetch next A chunk (global_prefetch_b8)
      __builtin_prefetch(Ar + (size_t)mrow * K + k0 + 32, 0, 1);
      __builtin_prefetch(Ai + (size_t)mrow * K + k0 + 32, 0, 1);
    }
    v16h ar  = a_frag_f32(Ar, K, mrow, k0, h);
    v16h ai  = a_frag_f32(Ai, K, mrow, k0, h);
    v16h ain = neg16(ai);
    v16h wr  = b_frag_f16(Wr, K, ncol, k0, h);
    v16h wi  = b_frag_f16(Wi, K, ncol, k0, h);
    accr = wmma16(ar,  wr, accr);   // + Ar*Wr
    accr = wmma16(ain, wi, accr);   // - Ai*Wi
    acci = wmma16(ar,  wi, acci);   // + Ar*Wi
    acci = wmma16(ai,  wr, acci);   // + Ai*Wr
  }
  const float bre = br[ncol];
  const float bie = bi[ncol];
  const int  pcol = ncol % kD;
#pragma unroll
  for (int r = 0; r < 8; ++r) {
    const int row = m0 + r + 8 * h;
    float vr = accr[r] + bre;
    float vi = acci[r] + bie;
    if (per) {
      vr += per[(size_t)row * kD + pcol];
      vi += pei[(size_t)row * kD + pcol];
    }
    if (mode == 0) {
      Orf[(size_t)row * N + ncol] = vr;
      Oif[(size_t)row * N + ncol] = vi;
    } else if (mode == 1) {
      Orh[(size_t)row * N + ncol] = (_Float16)vr;
      Oih[(size_t)row * N + ncol] = (_Float16)vi;
    } else {                       // transposed per-bh: [bh, N, S]
      const int bh = row >> 10;    // row / kS
      const int s  = row & (kS - 1);
      Orh[((size_t)bh * N + ncol) * kS + s] = (_Float16)vr;
      Oih[((size_t)bh * N + ncol) * kS + s] = (_Float16)vi;
    }
  }
}

// ---------------------------------------------------------------------------
// Fused complex differential attention. grid = (S/16, B*H), block = 128.
// Per 16-row q tile and score set: async-stage q tile into LDS, complex
// scores via WMMA -> magnitude*scale -> LDS f16 -> softmax -> probs @ V.
// LDS: 32KB scores + 24KB q tile + reductions (~57KB of 320KB/WGP).
// ---------------------------------------------------------------------------
__global__ void attn_kernel(const _Float16* __restrict__ qph_r,  // [M, 768]
                            const _Float16* __restrict__ qph_i,
                            const _Float16* __restrict__ kph_r,  // [M, 384]
                            const _Float16* __restrict__ kph_i,
                            const _Float16* __restrict__ vT_r,   // [bh, 384, S]
                            const _Float16* __restrict__ vT_i,
                            float* __restrict__ acat_r,          // [M, 768]
                            float* __restrict__ acat_i) {
  __shared__ _Float16 sc[16][kS];     // 32 KB score/prob tile
  __shared__ _Float16 sq_r[16][kD];   // 12 KB q tile (real)
  __shared__ _Float16 sq_i[16][kD];   // 12 KB q tile (imag)
  __shared__ float red[16][8];
  __shared__ float rowmax[16];
  __shared__ float rowinv[16];

  const int bh   = blockIdx.y;
  const int q0   = blockIdx.x * 16;
  const int wave = threadIdx.x >> 5;
  const int lane = threadIdx.x & 31;
  const int h    = lane >> 4;
  const int l15  = lane & 15;

  const _Float16* kbase_r = kph_r + (size_t)bh * kS * kD;
  const _Float16* kbase_i = kph_i + (size_t)bh * kS * kD;
  const _Float16* vbase_r = vT_r + (size_t)bh * kD * kS;
  const _Float16* vbase_i = vT_i + (size_t)bh * kD * kS;

  for (int set = 0; set < 2; ++set) {
    // ---- stage q tile (16 x 384 f16, r & i) into LDS asynchronously ----
    // 16 rows * 768 B/row = 48 x 16B chunks per row, 768 chunks per tensor.
    for (int c = threadIdx.x; c < 768; c += 128) {
      const int row  = c / 48;
      const int col8 = (c % 48) * 8;                  // f16 element offset
      const size_t goff =
          (size_t)(bh * kS + q0 + row) * kD2 + set * kD + col8;
      async_copy_b128((unsigned)(size_t)&sq_r[row][col8], qph_r + goff);
      async_copy_b128((unsigned)(size_t)&sq_i[row][col8], qph_i + goff);
    }
    wait_asynccnt0();
    __syncthreads();

    // ---- phase A: complex score magnitudes for all 1024 keys ----
    for (int kt = wave; kt < kS / 16; kt += 4) {
      const int kcol = kt * 16 + l15;
      v8f sr = {};
      v8f si = {};
      for (int k0 = 0; k0 < kD; k0 += 32) {
        v16h qr  = a_frag_f16(&sq_r[0][0], kD, l15, k0, h);   // ds_load
        v16h qi  = a_frag_f16(&sq_i[0][0], kD, l15, k0, h);   // ds_load
        v16h qrn = neg16(qr);
        v16h kr  = b_frag_f16(kbase_r, kD, kcol, k0, h);
        v16h ki  = b_frag_f16(kbase_i, kD, kcol, k0, h);
        sr = wmma16(qr,  kr, sr);   // + qr*kr
        sr = wmma16(qi,  ki, sr);   // + qi*ki
        si = wmma16(qi,  kr, si);   // + qi*kr
        si = wmma16(qrn, ki, si);   // - qr*ki
      }
#pragma unroll
      for (int r = 0; r < 8; ++r) {
        float a = sr[r], b = si[r];
        float mag = sqrtf(a * a + b * b + 1e-8f) * kScale;
        sc[r + 8 * h][kcol] = (_Float16)mag;   // ds_store
      }
    }
    __syncthreads();

    // ---- phase B: softmax over keys (8 threads per row) ----
    {
      const int row  = threadIdx.x >> 3;
      const int part = threadIdx.x & 7;
      float mx = -1e30f;
      for (int c = part; c < kS; c += 8) mx = fmaxf(mx, (float)sc[row][c]);
      red[row][part] = mx;
      __syncthreads();
      if (part == 0) {
        float m = red[row][0];
#pragma unroll
        for (int j = 1; j < 8; ++j) m = fmaxf(m, red[row][j]);
        rowmax[row] = m;
      }
      __syncthreads();
      const float m = rowmax[row];
      float sum = 0.f;
      for (int c = part; c < kS; c += 8) {
        float p = __expf((float)sc[row][c] - m);
        sc[row][c] = (_Float16)p;              // unnormalized probs
        sum += p;
      }
      red[row][part] = sum;
      __syncthreads();
      if (part == 0) {
        float t = 0.f;
#pragma unroll
        for (int j = 0; j < 8; ++j) t += red[row][j];
        rowinv[row] = 1.f / t;                 // fold 1/sum into the output
      }
      __syncthreads();
    }

    // ---- phase C: probs @ V (A from LDS, B from transposed V) ----
    for (int dt = wave; dt < kD / 16; dt += 4) {
      const int dcol = dt * 16 + l15;
      v8f ar = {};
      v8f ai = {};
      for (int k0 = 0; k0 < kS; k0 += 32) {
        if (k0 + 32 < kS) {
          __builtin_prefetch(vbase_r + (size_t)dcol * kS + k0 + 32, 0, 1);
          __builtin_prefetch(vbase_i + (size_t)dcol * kS + k0 + 32, 0, 1);
        }
        const _Float16* p = &sc[l15][k0 + 8 * h];
        v8h lo = *(const v8h*)p;               // ds_load_b128
        v8h hi = *(const v8h*)(p + 16);
        v16h pa;
#pragma unroll
        for (int j = 0; j < 8; ++j) { pa[j] = lo[j]; pa[8 + j] = hi[j]; }
        v16h vr = b_frag_f16(vbase_r, kS, dcol, k0, h);
        v16h vi = b_frag_f16(vbase_i, kS, dcol, k0, h);
        ar = wmma16(pa, vr, ar);
        ai = wmma16(pa, vi, ai);
      }
#pragma unroll
      for (int r = 0; r < 8; ++r) {
        const int lr    = r + 8 * h;
        const float inv = rowinv[lr];
        const size_t row = (size_t)(bh * kS + q0 + lr);
        acat_r[row * kD2 + set * kD + dcol] = ar[r] * inv;
        acat_i[row * kD2 + set * kD + dcol] = ai[r] * inv;
      }
    }
    __syncthreads();   // before next set reuses sc / sq
  }
}

// ---------------------------------------------------------------------------
// Complex RMSNorm (SubLN) + lambda combine + complex gating. grid = M, blk 256.
// ---------------------------------------------------------------------------
__global__ void norm_gate_kernel(const float* __restrict__ acat_r,
                                 const float* __restrict__ acat_i,
                                 const float* __restrict__ g_r,
                                 const float* __restrict__ g_i,
                                 const float* __restrict__ subln_w,
                                 const float* __restrict__ lam_p,
                                 float* __restrict__ go_r,
                                 float* __restrict__ go_i) {
  __shared__ float ssum[256];
  const int m = blockIdx.x;
  const int t = threadIdx.x;
  const float* cr = acat_r + (size_t)m * kD2;
  const float* ci = acat_i + (size_t)m * kD2;
  float local = 0.f;
  for (int d = t; d < kD2; d += 256) {
    float a = cr[d], b = ci[d];
    local += a * a + b * b;
  }
  ssum[t] = local;
  __syncthreads();
  for (int s = 128; s > 0; s >>= 1) {
    if (t < s) ssum[t] += ssum[t + s];
    __syncthreads();
  }
  const float inv = 1.0f / sqrtf(ssum[0] / (float)kD2 + 1e-5f);
  const float lam = lam_p[0];
  for (int d = t; d < kD; d += 256) {
    float w1  = subln_w[d], w2 = subln_w[d + kD];
    float a1r = cr[d] * inv * w1,        a1i = ci[d] * inv * w1;
    float a2r = cr[d + kD] * inv * w2,   a2i = ci[d + kD] * inv * w2;
    float outr = a1r - lam * a2r;
    float outi = a1i - lam * a2i;
    float gr = g_r[(size_t)m * kD + d];
    float gi = g_i[(size_t)m * kD + d];
    go_r[(size_t)m * kD + d] = gr * outr - gi * outi;
    go_i[(size_t)m * kD + d] = gr * outi + gi * outr;
  }
}

__global__ void lambda_kernel(const float* __restrict__ lq1,
                              const float* __restrict__ lk1,
                              const float* __restrict__ lq2,
                              const float* __restrict__ lk2,
                              float* __restrict__ out) {
  __shared__ float s1[kD], s2[kD];
  const int t = threadIdx.x;
  s1[t] = lq1[t] * lk1[t];
  s2[t] = lq2[t] * lk2[t];
  __syncthreads();
  if (t == 0) {
    float a = 0.f, b = 0.f;
    for (int j = 0; j < kD; ++j) { a += s1[j]; b += s2[j]; }
    float x = __expf(a) - __expf(b) + kLambdaInit;
    out[0] = 1.f / (1.f + __expf(-x));
  }
}

__global__ void f32_to_h_kernel(const float* __restrict__ s,
                                _Float16* __restrict__ d, int n) {
  int i = blockIdx.x * 256 + threadIdx.x;
  if (i < n) d[i] = (_Float16)s[i];
}

// ---------------------------------------------------------------------------
extern "C" void kernel_launch(void* const* d_in, const int* in_sizes, int n_in,
                              void* d_out, int out_size, void* d_ws,
                              size_t ws_size, hipStream_t stream) {
  (void)in_sizes; (void)n_in; (void)out_size; (void)ws_size;

  const float* q_r   = (const float*)d_in[0];
  const float* q_i   = (const float*)d_in[1];
  const float* k_r   = (const float*)d_in[2];
  const float* k_i   = (const float*)d_in[3];
  const float* v_r   = (const float*)d_in[4];
  const float* v_i   = (const float*)d_in[5];
  const float* pe_q_r = (const float*)d_in[6];
  const float* pe_q_i = (const float*)d_in[7];
  const float* pe_k_r = (const float*)d_in[8];
  const float* pe_k_i = (const float*)d_in[9];
  const float* wq_r = (const float*)d_in[10];
  const float* wq_i = (const float*)d_in[11];
  const float* bq_r = (const float*)d_in[12];
  const float* bq_i = (const float*)d_in[13];
  const float* wk_r = (const float*)d_in[14];
  const float* wk_i = (const float*)d_in[15];
  const float* bk_r = (const float*)d_in[16];
  const float* bk_i = (const float*)d_in[17];
  const float* wv_r = (const float*)d_in[18];
  const float* wv_i = (const float*)d_in[19];
  const float* bv_r = (const float*)d_in[20];
  const float* bv_i = (const float*)d_in[21];
  const float* wg_r = (const float*)d_in[22];
  const float* wg_i = (const float*)d_in[23];
  const float* bg_r = (const float*)d_in[24];
  const float* bg_i = (const float*)d_in[25];
  const float* wo_r = (const float*)d_in[26];
  const float* wo_i = (const float*)d_in[27];
  const float* bo_r = (const float*)d_in[28];
  const float* bo_i = (const float*)d_in[29];
  const float* lam_q1 = (const float*)d_in[30];
  const float* lam_k1 = (const float*)d_in[31];
  const float* lam_q2 = (const float*)d_in[32];
  const float* lam_k2 = (const float*)d_in[33];
  const float* subln_w = (const float*)d_in[34];

  // --- workspace carve-up (bump allocator, 256B aligned) ---
  char* ws = (char*)d_ws;
  auto alloc = [&](size_t bytes) -> char* {
    char* r = ws;
    ws += (bytes + 255) & ~(size_t)255;
    return r;
  };
  _Float16* wqh_r = (_Float16*)alloc((size_t)kD2 * kD * 2);
  _Float16* wqh_i = (_Float16*)alloc((size_t)kD2 * kD * 2);
  _Float16* wkh_r = (_Float16*)alloc((size_t)kD * kD * 2);
  _Float16* wkh_i = (_Float16*)alloc((size_t)kD * kD * 2);
  _Float16* wvh_r = (_Float16*)alloc((size_t)kD * kD * 2);
  _Float16* wvh_i = (_Float16*)alloc((size_t)kD * kD * 2);
  _Float16* wgh_r = (_Float16*)alloc((size_t)kD * kD * 2);
  _Float16* wgh_i = (_Float16*)alloc((size_t)kD * kD * 2);
  _Float16* woh_r = (_Float16*)alloc((size_t)kD * kD * 2);
  _Float16* woh_i = (_Float16*)alloc((size_t)kD * kD * 2);
  _Float16* qph_r = (_Float16*)alloc((size_t)kM * kD2 * 2);
  _Float16* qph_i = (_Float16*)alloc((size_t)kM * kD2 * 2);
  _Float16* kph_r = (_Float16*)alloc((size_t)kM * kD * 2);
  _Float16* kph_i = (_Float16*)alloc((size_t)kM * kD * 2);
  _Float16* vT_r  = (_Float16*)alloc((size_t)kM * kD * 2);
  _Float16* vT_i  = (_Float16*)alloc((size_t)kM * kD * 2);
  float* gws_r  = (float*)alloc((size_t)kM * kD * 4);
  float* gws_i  = (float*)alloc((size_t)kM * kD * 4);
  float* acat_r = (float*)alloc((size_t)kM * kD2 * 4);
  float* acat_i = (float*)alloc((size_t)kM * kD2 * 4);
  float* go_r   = (float*)alloc((size_t)kM * kD * 4);
  float* go_i   = (float*)alloc((size_t)kM * kD * 4);
  float* lam_ws = (float*)alloc(256);

  // --- weight conversion f32 -> f16 ---
  auto conv = [&](const float* src, _Float16* dst, int n) {
    f32_to_h_kernel<<<dim3((n + 255) / 256), dim3(256), 0, stream>>>(src, dst, n);
  };
  conv(wq_r, wqh_r, kD2 * kD);  conv(wq_i, wqh_i, kD2 * kD);
  conv(wk_r, wkh_r, kD * kD);   conv(wk_i, wkh_i, kD * kD);
  conv(wv_r, wvh_r, kD * kD);   conv(wv_i, wvh_i, kD * kD);
  conv(wg_r, wgh_r, kD * kD);   conv(wg_i, wgh_i, kD * kD);
  conv(wo_r, woh_r, kD * kD);   conv(wo_i, woh_i, kD * kD);

  lambda_kernel<<<dim3(1), dim3(kD), 0, stream>>>(lam_q1, lam_k1, lam_q2,
                                                  lam_k2, lam_ws);

  const dim3 blk(128);
  // q projection -> f16 [M, 768] with fused PE concat-add
  cgemm_kernel<<<dim3(kM / 16, kD2 / 64), blk, 0, stream>>>(
      q_r, q_i, wqh_r, wqh_i, bq_r, bq_i, kD2, kD, pe_q_r, pe_q_i, 1,
      nullptr, nullptr, qph_r, qph_i);
  // k projection -> f16 [M, 384] with fused PE add
  cgemm_kernel<<<dim3(kM / 16, kD / 64), blk, 0, stream>>>(
      k_r, k_i, wkh_r, wkh_i, bk_r, bk_i, kD, kD, pe_k_r, pe_k_i, 1,
      nullptr, nullptr, kph_r, kph_i);
  // v projection -> f16 transposed [bh, 384, S]
  cgemm_kernel<<<dim3(kM / 16, kD / 64), blk, 0, stream>>>(
      v_r, v_i, wvh_r, wvh_i, bv_r, bv_i, kD, kD, nullptr, nullptr, 2,
      nullptr, nullptr, vT_r, vT_i);
  // gate projection -> f32 [M, 384]
  cgemm_kernel<<<dim3(kM / 16, kD / 64), blk, 0, stream>>>(
      q_r, q_i, wgh_r, wgh_i, bg_r, bg_i, kD, kD, nullptr, nullptr, 0,
      gws_r, gws_i, nullptr, nullptr);

  // fused differential attention
  attn_kernel<<<dim3(kS / 16, kB * kH), dim3(128), 0, stream>>>(
      qph_r, qph_i, kph_r, kph_i, vT_r, vT_i, acat_r, acat_i);

  // SubLN + lambda + gating
  norm_gate_kernel<<<dim3(kM), dim3(256), 0, stream>>>(
      acat_r, acat_i, gws_r, gws_i, subln_w, lam_ws, go_r, go_i);

  // output projection straight into d_out (out_r then out_i, flat)
  float* out_r = (float*)d_out;
  float* out_i = out_r + (size_t)kM * kD;
  cgemm_kernel<<<dim3(kM / 16, kD / 64), blk, 0, stream>>>(
      go_r, go_i, woh_r, woh_i, bo_r, bo_i, kD, kD, nullptr, nullptr, 0,
      out_r, out_i, nullptr, nullptr);
}
